// MelSpectrogram_19954418057364
// MI455X (gfx1250) — matrix operand order repaired and checked by reading, not compiled, not run on hardware
//
#include <hip/hip_runtime.h>
#include <hip/hip_bf16.h>
#include <math.h>

// ---------------- problem constants ----------------
#define SR        16000
#define WIN       160
#define HOP       80
#define NFFT      512
#define FBINS     257          // NFFT/2+1
#define NFILT     40
#define BATCH     32
#define SAMPLES   160160
#define TBINS     2001         // (SAMPLES-WIN)/HOP + 1
#define FRAMES    (BATCH*TBINS)          // 64032 = 16*4002
#define MTILES    (FRAMES/16)            // 4002
#define MPAIRS    (MTILES/2)             // 2001 (each wave owns 2 M-tiles)
#define NT1       17           // freq tiles: ceil(257/16) -> pad to 272
#define KB1       5            // K steps over WIN=160 (32 each)
#define NT2       3            // mel tiles: ceil(40/16) -> pad to 48
#define KB2       9            // K steps over 272 freqs -> pad to 288
#define EPSF      2.220446049250313e-16f
#define PI_D      3.14159265358979323846

// packed fragment tables in workspace (f16), 512 halves (=32 lanes x 16) per fragment
#define TW_FRAGS  (2*NT1*KB1)            // cos tiles then sin tiles: 170
#define BK_FRAGS  (NT2*KB2)              // 27
#define TW_HALVES (TW_FRAGS*512)         // 87040
#define BK_HALVES (BK_FRAGS*512)         // 13824

typedef __attribute__((ext_vector_type(16))) _Float16 v16h;
typedef __attribute__((ext_vector_type(8)))  float    v8f;

union Frag { v16h h; uint4 u[2]; };

// =====================================================================
// Setup kernel 1: packed twiddle table.
// B-fragment layout (16-bit B, 32x16): lane&15 = column n, lane>>4 = K half,
// halves i=0..15 hold k = kb*32 + (lane>>4)*8 + (i<8 ? i : i+8).
// Fragments 0..84 = cos, 85..169 = sin; column n >= 257 is zero padding.
// =====================================================================
__global__ void mel_tw_pack(_Float16* __restrict__ tw) {
    int tid = blockIdx.x * blockDim.x + threadIdx.x;
    if (tid >= TW_HALVES) return;
    int i    = tid & 15;
    int lane = (tid >> 4) & 31;
    int fi   = tid >> 9;                 // fragment index = nt*KB1 + kb
    int kb   = fi % KB1;
    int nt   = fi / KB1;                 // 0..33
    int n    = (nt % NT1) * 16 + (lane & 15);
    int k    = kb * 32 + ((lane >> 4) * 8) + (i < 8 ? i : i + 8);   // 0..159
    float val = 0.0f;
    if (n < FBINS) {
        int   kn = (k * n) & (NFFT - 1);           // exact mod-512 phase reduction
        float th = (float)kn * (float)(PI_D / 256.0);
        val = (nt < NT1) ? cosf(th) : sinf(th);    // sign of imag part irrelevant (squared)
    }
    tw[tid] = (_Float16)val;
}

// =====================================================================
// Setup kernel 2: packed mel filterbank, transposed to (freq x filter),
// same B-fragment layout: lane&15 = filter j, halves hold freq k.
// (The reference's offset loop is identity since idx[0]==0.)
// =====================================================================
__device__ inline int mel_idx(int j) {
    double mel_max = 2595.0 * log10(1.0 + (double)SR / 700.0);
    double m  = mel_max * (double)j / 41.0;        // linspace(0, mel_max, 42)
    double hz = 700.0 * (pow(10.0, m / 2595.0) - 1.0);
    return (int)(hz * (double)FBINS / (double)SR);
}

__global__ void mel_bank_pack(_Float16* __restrict__ bk) {
    int tid = blockIdx.x * blockDim.x + threadIdx.x;
    if (tid >= BK_HALVES) return;
    int i    = tid & 15;
    int lane = (tid >> 4) & 31;
    int fi   = tid >> 9;                 // nt2*KB2 + kb2
    int kb2  = fi % KB2;
    int nt2  = fi / KB2;
    int j    = nt2 * 16 + (lane & 15);   // mel filter
    int k    = kb2 * 32 + ((lane >> 4) * 8) + (i < 8 ? i : i + 8);  // freq (0..287)
    float val = 0.0f;
    if (j < NFILT && k < FBINS) {
        int l = mel_idx(j), m = mel_idx(j + 1), r = mel_idx(j + 2);
        if (k >= l && k < m && m > l)       val = (float)(k - l) / (float)(m - l);
        else if (k >= m && k < r && r > m)  val = 1.0f - (float)(k - m) / (float)(r - m);
    }
    bk[tid] = (_Float16)val;
}

// =====================================================================
// Main fused kernel: frames -> |rfft|^2/NFFT -> mel -> log.
// Register-blocked: each wave owns MBLK=2 16-frame tiles so every twiddle /
// filterbank B fragment loaded from L2 feeds 2 WMMAs (halves L2 traffic,
// which roofline says is the dominant term). 2*(170+27) WMMAs per wave.
// =====================================================================
#define WPB  2    // waves per block (LDS budget: 2*2*16*296*2 = 37,888 B)
#define MBLK 2    // M-tiles per wave
#define LDSK 296  // padded row stride (halves): conflict-free b128 reads

__global__ __launch_bounds__(WPB * 32) void mel_main(
        const float* __restrict__ audio,
        const _Float16* __restrict__ tw,
        const _Float16* __restrict__ bk,
        float* __restrict__ out) {
    __shared__ __align__(16) _Float16 lds[WPB][MBLK][16][LDSK];

    const int lane = threadIdx.x & 31;
    const int wv   = threadIdx.x >> 5;
    const int pair = blockIdx.x * WPB + wv;
    if (pair >= MPAIRS) return;          // wave-uniform: EXEC stays all-ones
    const int tile0 = pair * MBLK;       // tiles tile0, tile0+1 (both < MTILES)

    const int nlo = lane & 15;           // column / row-within-tile selector
    const int khi = lane >> 4;           // K-half selector

    // ---- A fragments: MBLK x (16 frames x 160 samples), f32 -> f16 ----
    // A layout (16-bit A 16x32): lane&15 = M row; halves i: k = khi*8 + (i<8 ? i : i+8)
    Frag a[MBLK][KB1];
    #pragma unroll
    for (int mb = 0; mb < MBLK; ++mb) {
        const int f = (tile0 + mb) * 16 + nlo;   // flat frame index for this lane's row
        const int b = f / TBINS;
        const int t = f - b * TBINS;
        const float* src = audio + (size_t)b * SAMPLES + (size_t)t * HOP;
        #pragma unroll
        for (int kb = 0; kb < KB1; ++kb) {
            const int base = kb * 32 + khi * 8;  // 32B-aligned offsets
            float4 r0 = *(const float4*)(src + base);
            float4 r1 = *(const float4*)(src + base + 4);
            float4 r2 = *(const float4*)(src + base + 16);
            float4 r3 = *(const float4*)(src + base + 20);
            v16h h;
            h[0]=(_Float16)r0.x; h[1]=(_Float16)r0.y; h[2]=(_Float16)r0.z; h[3]=(_Float16)r0.w;
            h[4]=(_Float16)r1.x; h[5]=(_Float16)r1.y; h[6]=(_Float16)r1.z; h[7]=(_Float16)r1.w;
            h[8]=(_Float16)r2.x; h[9]=(_Float16)r2.y; h[10]=(_Float16)r2.z; h[11]=(_Float16)r2.w;
            h[12]=(_Float16)r3.x; h[13]=(_Float16)r3.y; h[14]=(_Float16)r3.z; h[15]=(_Float16)r3.w;
            a[mb][kb].h = h;
        }
    }

    const uint4* twv = (const uint4*)tw;  // 2 uint4 per lane per fragment
    const uint4* bkv = (const uint4*)bk;

    // ---- Stage 1: DFT power spectrum, 17 freq tiles; B frags shared by both M tiles ----
    for (int nt = 0; nt < NT1; ++nt) {
        v8f accR[MBLK], accI[MBLK];
        #pragma unroll
        for (int mb = 0; mb < MBLK; ++mb) {
            accR[mb] = (v8f){0.f,0.f,0.f,0.f,0.f,0.f,0.f,0.f};
            accI[mb] = (v8f){0.f,0.f,0.f,0.f,0.f,0.f,0.f,0.f};
        }
        #pragma unroll
        for (int kb = 0; kb < KB1; ++kb) {
            const int fc = nt * KB1 + kb;              // cos fragment
            const int fs = (nt + NT1) * KB1 + kb;      // sin fragment
            Frag bc, bs;
            bc.u[0] = twv[fc * 64 + lane * 2];
            bc.u[1] = twv[fc * 64 + lane * 2 + 1];
            bs.u[0] = twv[fs * 64 + lane * 2];
            bs.u[1] = twv[fs * 64 + lane * 2 + 1];
            #pragma unroll
            for (int mb = 0; mb < MBLK; ++mb) {
                accR[mb] = __builtin_amdgcn_wmma_f32_16x16x32_f16(
                               false, a[mb][kb].h, false, bc.h, (short)0, accR[mb], false, false);
                accI[mb] = __builtin_amdgcn_wmma_f32_16x16x32_f16(
                               false, a[mb][kb].h, false, bs.h, (short)0, accI[mb], false, false);
            }
        }
        // C/D layout: VGPR v, lane -> (M = v + 8*khi, N = nlo). P = (R^2+I^2)/NFFT.
        #pragma unroll
        for (int mb = 0; mb < MBLK; ++mb) {
            #pragma unroll
            for (int v = 0; v < 8; ++v) {
                float p = fmaf(accR[mb][v], accR[mb][v], accI[mb][v] * accI[mb][v])
                          * (1.0f / (float)NFFT);
                lds[wv][mb][v + 8 * khi][nt * 16 + nlo] = (_Float16)p;
            }
        }
    }
    // zero the K-pad region (freq 272..287) so stage-2 K steps see zeros
    {
        uint4 z; z.x = 0; z.y = 0; z.z = 0; z.w = 0;
        #pragma unroll
        for (int mb = 0; mb < MBLK; ++mb)
            *(uint4*)&lds[wv][mb][nlo][272 + khi * 8] = z;
    }
    asm volatile("s_wait_dscnt 0" ::: "memory");   // LDS write->read turn-around (same wave)

    // ---- Stage 2: mel projection + log; filterbank frag shared by both M tiles ----
    for (int nt2 = 0; nt2 < NT2; ++nt2) {
        v8f acc[MBLK];
        #pragma unroll
        for (int mb = 0; mb < MBLK; ++mb)
            acc[mb] = (v8f){0.f,0.f,0.f,0.f,0.f,0.f,0.f,0.f};
        #pragma unroll
        for (int kb2 = 0; kb2 < KB2; ++kb2) {
            const int base = kb2 * 32 + khi * 8;   // A-layout K runs [base,+8),[base+16,+8)
            const int fb = nt2 * KB2 + kb2;
            Frag b2;
            b2.u[0] = bkv[fb * 64 + lane * 2];
            b2.u[1] = bkv[fb * 64 + lane * 2 + 1];
            #pragma unroll
            for (int mb = 0; mb < MBLK; ++mb) {
                Frag a2;
                a2.u[0] = *(const uint4*)&lds[wv][mb][nlo][base];
                a2.u[1] = *(const uint4*)&lds[wv][mb][nlo][base + 16];
                acc[mb] = __builtin_amdgcn_wmma_f32_16x16x32_f16(
                              false, a2.h, false, b2.h, (short)0, acc[mb], false, false);
            }
        }
        const int n = nt2 * 16 + nlo;              // mel filter index
        if (n < NFILT) {
            #pragma unroll
            for (int mb = 0; mb < MBLK; ++mb) {
                #pragma unroll
                for (int v = 0; v < 8; ++v) {
                    const int row = v + 8 * khi;   // frame within tile
                    float x = acc[mb][v];
                    x = fmaxf(x, EPSF);
                    out[(size_t)((tile0 + mb) * 16 + row) * NFILT + n] = logf(x);
                }
            }
        }
    }
}

// =====================================================================
extern "C" void kernel_launch(void* const* d_in, const int* in_sizes, int n_in,
                              void* d_out, int out_size, void* d_ws, size_t ws_size,
                              hipStream_t stream) {
    const float* audio = (const float*)d_in[0];
    float* out = (float*)d_out;
    _Float16* tw = (_Float16*)d_ws;             // 174,080 B
    _Float16* bk = tw + TW_HALVES;              // +27,648 B  (total < 256 KB of ws)

    mel_tw_pack  <<<(TW_HALVES + 255) / 256, 256, 0, stream>>>(tw);
    mel_bank_pack<<<(BK_HALVES + 255) / 256, 256, 0, stream>>>(bk);

    const int blocks = (MPAIRS + WPB - 1) / WPB;   // 1001
    mel_main<<<blocks, WPB * 32, 0, stream>>>(audio, tw, bk, out);
}
